// AnchorPositionalEncoding_62612033241328
// MI455X (gfx1250) — matrix operand order repaired and checked by reading, not compiled, not run on hardware
//
#include <hip/hip_runtime.h>
#include <hip/hip_bf16.h>

typedef __attribute__((ext_vector_type(2)))  float    v2f;
typedef __attribute__((ext_vector_type(8)))  float    v8f;
typedef __attribute__((ext_vector_type(16))) _Float16 v16h;

#define N_NODES 16384
#define AN 64      // NUM_ANCHORS
#define HD 128     // HIDDEN_DIM

// ---------------------------------------------------------------------------
// Stage 1: deg[row] = sum(adj[row, :]).  1 GiB streaming read -> HBM bound.
// One 256-thread block per row (64 KB contiguous), float4 loads, wave32 reduce.
// ---------------------------------------------------------------------------
__global__ void __launch_bounds__(256)
deg_kernel(const float* __restrict__ adj, float* __restrict__ deg) {
    const int row  = blockIdx.x;
    const int lane = threadIdx.x & 31;
    const int wv   = threadIdx.x >> 5;

    const float4* rp = (const float4*)(adj + (size_t)row * N_NODES);
    float s = 0.0f;
    #pragma unroll 4
    for (int i = threadIdx.x; i < N_NODES / 4; i += 256) {
        float4 v = rp[i];
        s += (v.x + v.y) + (v.z + v.w);
    }
    #pragma unroll
    for (int off = 16; off > 0; off >>= 1) s += __shfl_xor(s, off, 32);

    __shared__ float red[8];
    if (lane == 0) red[wv] = s;
    __syncthreads();
    if (wv == 0) {
        s = (lane < 8) ? red[lane] : 0.0f;
        #pragma unroll
        for (int off = 4; off > 0; off >>= 1) s += __shfl_xor(s, off, 32);
        if (lane == 0) deg[row] = s;
    }
}

// ---------------------------------------------------------------------------
// Stage 2: nrm = sqrt(sum(deg^2)) + 1e-6.  Single block, 64 KB read: trivial.
// ---------------------------------------------------------------------------
__global__ void __launch_bounds__(256)
nrm_kernel(const float* __restrict__ deg, float* __restrict__ nrm_out) {
    const int lane = threadIdx.x & 31;
    const int wv   = threadIdx.x >> 5;
    float s = 0.0f;
    for (int i = threadIdx.x; i < N_NODES; i += 256) {
        float d = deg[i];
        s += d * d;
    }
    #pragma unroll
    for (int off = 16; off > 0; off >>= 1) s += __shfl_xor(s, off, 32);
    __shared__ float red[8];
    if (lane == 0) red[wv] = s;
    __syncthreads();
    if (wv == 0) {
        s = (lane < 8) ? red[lane] : 0.0f;
        #pragma unroll
        for (int off = 4; off > 0; off >>= 1) s += __shfl_xor(s, off, 32);
        if (lane == 0) nrm_out[0] = sqrtf(s) + 1e-6f;
    }
}

// ---------------------------------------------------------------------------
// Stage 3: fused  sim -> softmax -> (W @ anchor_emb)  via WMMA.
// Block = 16 output rows x 128 cols. 8 waves: softmax (2 rows/wave) into LDS,
// then each wave owns one 16-wide N-tile and runs K=64 as chained WMMAs.
// ---------------------------------------------------------------------------
__global__ void __launch_bounds__(256)
out_kernel(const float* __restrict__ deg, const float* __restrict__ nrm_p,
           const float* __restrict__ E, float* __restrict__ out) {
    __shared__ float sW[16][AN];   // softmax weights tile, 4 KB
    __shared__ float sE[AN][HD];   // anchor_emb,           32 KB

    const int tid  = threadIdx.x;
    const int lane = tid & 31;
    const int wave = tid >> 5;          // 0..7
    const int m0   = blockIdx.x * 16;   // first output row of this block

    // Stage E (64x128 f32) into LDS with float4 loads: 8 per thread.
    {
        const float4* e4  = (const float4*)E;
        float4*       se4 = (float4*)&sE[0][0];
        #pragma unroll
        for (int i = tid; i < AN * HD / 4; i += 256) se4[i] = e4[i];
    }

    // Softmax of sim[i, a] = deg[i]*deg[a]/nrm over a (64 anchors).
    // Wave handles rows 2*wave, 2*wave+1; lane covers anchors {lane, lane+32}.
    const float inv_nrm = 1.0f / nrm_p[0];
    const float dA0 = deg[lane];
    const float dA1 = deg[lane + 32];
    #pragma unroll
    for (int rr = 0; rr < 2; ++rr) {
        const int   r = 2 * wave + rr;
        const float c = deg[m0 + r] * inv_nrm;
        float v0 = c * dA0;
        float v1 = c * dA1;
        float mx = fmaxf(v0, v1);
        #pragma unroll
        for (int off = 16; off > 0; off >>= 1) mx = fmaxf(mx, __shfl_xor(mx, off, 32));
        float e0 = __expf(v0 - mx);
        float e1 = __expf(v1 - mx);
        float sum = e0 + e1;
        #pragma unroll
        for (int off = 16; off > 0; off >>= 1) sum += __shfl_xor(sum, off, 32);
        const float inv = 1.0f / sum;
        sW[r][lane]      = e0 * inv;
        sW[r][lane + 32] = e1 * inv;
    }
    __syncthreads();

    // WMMA: wave 'w' computes out[m0:m0+16, 16w:16w+16].
    const int nt   = wave * 16;
    const int mrow = lane & 15;
    const int col  = nt + (lane & 15);
    v8f acc = {};

#if __has_builtin(__builtin_amdgcn_wmma_f32_16x16x4_f32)
    // Full-f32 path: V_WMMA_F32_16X16X4_F32, K=64 as 16 chained k-steps.
    // A 16x4 f32 layout: lanes 0-15 -> K {0,1}; lanes 16-31 -> K {2,3} (M = lane&15).
    // B  4x16 f32 layout: VGPR0 -> K {0 | 2}, VGPR1 -> K {1 | 3}, N = lane&15.
    const int koff = (lane < 16) ? 0 : 2;
    #pragma unroll
    for (int k0 = 0; k0 < AN; k0 += 4) {
        v2f a, b;
        a.x = sW[mrow][k0 + koff];
        a.y = sW[mrow][k0 + koff + 1];
        b.x = sE[k0 + koff][col];
        b.y = sE[k0 + koff + 1][col];
        acc = __builtin_amdgcn_wmma_f32_16x16x4_f32(
                  false, a, false, b, (short)0, acc, false, false);
    }
#else
    // Fallback (codegen-confirmed): f16 inputs, f32 accumulate, K=64 = 2 x 32.
    const int h = lane >> 4;
    #pragma unroll
    for (int k0 = 0; k0 < AN; k0 += 32) {
        v16h a, b;
        #pragma unroll
        for (int e = 0; e < 16; ++e) {
            const int j = e >> 1, p = e & 1;
            const int ka = ((j < 4) ? 0 : 16) + 8 * h + 2 * (j & 3) + p; // ISA 16-bit A 16x32 layout
            const int kb = ((e < 8) ? 0 : 16) + 8 * h + (e & 7);         // B 32x16, same k order as A
            a[e] = (_Float16)sW[mrow][k0 + ka];
            b[e] = (_Float16)sE[k0 + kb][col];
        }
        acc = __builtin_amdgcn_wmma_f32_16x16x32_f16(
                  false, a, false, b, (short)0, acc, false, false);
    }
#endif

    // C/D 16x16 f32 layout: VGPR r -> M = r (lanes 0-15) / r+8 (lanes 16-31).
    const int mres = (lane < 16) ? 0 : 8;
    #pragma unroll
    for (int r = 0; r < 8; ++r) {
        out[(size_t)(m0 + r + mres) * HD + col] = acc[r];
    }
}

// ---------------------------------------------------------------------------
extern "C" void kernel_launch(void* const* d_in, const int* in_sizes, int n_in,
                              void* d_out, int out_size, void* d_ws, size_t ws_size,
                              hipStream_t stream) {
    const float* adj = (const float*)d_in[0];   // [16384, 16384] f32
    const float* emb = (const float*)d_in[1];   // [64, 128] f32
    float*       out = (float*)d_out;           // [16384, 128] f32

    float* deg = (float*)d_ws;                  // 16384 floats
    float* nrm = deg + N_NODES;                 // 1 float

    deg_kernel<<<N_NODES, 256, 0, stream>>>(adj, deg);
    nrm_kernel<<<1, 256, 0, stream>>>(deg, nrm);
    out_kernel<<<N_NODES / 16, 256, 0, stream>>>(deg, nrm, emb, out);
}
